// Model_61856118997672
// MI455X (gfx1250) — matrix operand order repaired and checked by reading, not compiled, not run on hardware
//
#include <hip/hip_runtime.h>

// ---------------------------------------------------------------------------
// Problem constants (from reference)
// ---------------------------------------------------------------------------
#define ISSUE_N 8192
#define DEV_N   4096
#define FILE_N  4096
#define N_TOT   (ISSUE_N + DEV_N + FILE_N)   // 16384
#define LATDIM  64
#define HYPER_D 128
#define LAYERS  2

typedef __attribute__((ext_vector_type(16))) __bf16    v16bf;
typedef __attribute__((ext_vector_type(8)))  float     v8f;
typedef __attribute__((ext_vector_type(4)))  float     f4;
typedef __attribute__((ext_vector_type(4)))  unsigned  u4;

union Frag {
    unsigned u[8];
    u4       q[2];
    v16bf    v;
};

// float pair -> packed bf16 dword.
// Round-half-up (+0x8000) then v_perm_b32 assembles {hi[31:16], lo[31:16]}:
// 3 VALU ops total per packed dword.
__device__ __forceinline__ unsigned pk2(float lo, float hi) {
    unsigned a = __float_as_uint(lo) + 0x8000u;
    unsigned b = __float_as_uint(hi) + 0x8000u;
    // v_perm_b32: byte codes 0-3 pick from arg1 (a), 4-7 from arg0 (b).
    // D = { b.byte3, b.byte2, a.byte3, a.byte2 }
    return __builtin_amdgcn_perm(b, a, 0x07060302u);
}
__device__ __forceinline__ float lrelu(float x) { return fmaxf(x, 0.5f * x); }

// ---------------------------------------------------------------------------
// Repack row-major fp32 [rows, cols] into WMMA B-layout bf16 fragments.
// Fragment (ktile, ntile): 32x16 block. dword v of lane l holds
//   (B[k0 + (v>>2)*16 + (l>>4)*8 + (v&3)*2, n0 + (l&15)], B[... + 1, ...]) packed bf16.
// Fragment buffer: [(ktile * (cols/16)) + ntile] * 256 + lane*8 + v
// ---------------------------------------------------------------------------
__global__ __launch_bounds__(256) void repack_b(const float* __restrict__ src,
                                                unsigned* __restrict__ dst,
                                                int rows, int cols) {
    int tid  = blockIdx.x * blockDim.x + threadIdx.x;
    int lane = tid & 31;
    int frag = tid >> 5;
    int ntiles = cols >> 4;
    int total  = (rows >> 5) * ntiles;
    if (frag >= total) return;
    int kt = frag / ntiles;
    int nt = frag - kt * ntiles;
    int n     = (nt << 4) + (lane & 15);
    int kbase = (kt << 5) + ((lane >> 4) << 3);
    unsigned* d = dst + ((size_t)frag << 8) + (lane << 3);
#pragma unroll
    for (int v = 0; v < 8; ++v) {
        int k = kbase + ((v >> 2) << 4) + ((v & 3) << 1);
        d[v] = pk2(src[(size_t)k * cols + n], src[(size_t)(k + 1) * cols + n]);
    }
}

// ---------------------------------------------------------------------------
// Repack H [nodes, 128] row-major fp32 into A-layout bf16 fragments of H^T
// (M = 128 -> 8 mtiles, K = nodes). A-fragment: dword v of lane l holds
//   (A[m0+(l&15), k], A[m0+(l&15), k+1]),  k = k0 + (v>>2)*16 + (l>>4)*8 + (v&3)*2
// Fragment buffer: [mtile * nk + ktile] * 256 + lane*8 + v
// ---------------------------------------------------------------------------
__global__ __launch_bounds__(256) void repack_at(const float* __restrict__ H,
                                                 unsigned* __restrict__ dst,
                                                 int nodes) {
    int tid  = blockIdx.x * blockDim.x + threadIdx.x;
    int lane = tid & 31;
    int frag = tid >> 5;
    int nk = nodes >> 5;
    if (frag >= 8 * nk) return;
    int mtile = frag / nk;
    int kt    = frag - mtile * nk;
    int m = (mtile << 4) + (lane & 15);
    unsigned* d = dst + ((size_t)frag << 8) + (lane << 3);
#pragma unroll
    for (int v = 0; v < 8; ++v) {
        int k = (kt << 5) + ((v >> 2) << 4) + ((lane >> 4) << 3) + ((v & 3) << 1);
        d[v] = pk2(H[(size_t)k * HYPER_D + m], H[(size_t)(k + 1) * HYPER_D + m]);
    }
}

// ---------------------------------------------------------------------------
// Generic GEMM: C[M, *] = act?( A_fp32[M,K] @ B ), B pre-swizzled bf16.
// One wave computes a 32x64 tile (8 WMMA accumulators; two A fragments share
// every B fragment). A is streamed from global fp32 with non-temporal b128
// loads (read-once; keep reused B resident in L2) and packed to bf16 with
// v_perm_b32. B fragments are 2x b128 loads per WMMA pair.
// Requires M % 32 == 0, K % 32 == 0. grid.y = 64-column block.
// ---------------------------------------------------------------------------
__global__ __launch_bounds__(256) void gemm_a32(const float* __restrict__ A,
                                                const unsigned* __restrict__ Bf,
                                                float* __restrict__ C,
                                                int M, int K, int lda, int ldc,
                                                int ntilesB, int act) {
    int wave = blockIdx.x * (blockDim.x >> 5) + (threadIdx.x >> 5);
    int lane = threadIdx.x & 31;
    if (wave * 32 >= M) return;
    int m0   = wave << 5;
    int cb   = blockIdx.y;
    int half = lane >> 4;
    int lm   = lane & 15;

    v8f acc[8] = {};
    const float* arow0 = A + (size_t)(m0 + lm) * lda + (half << 3);
    const float* arow1 = arow0 + (size_t)16 * lda;
    int nk = K >> 5;

    for (int kt = 0; kt < nk; ++kt) {
        const float* ap0 = arow0 + ((size_t)kt << 5);
        const float* ap1 = arow1 + ((size_t)kt << 5);
        __builtin_prefetch((const void*)(ap0 + 256), 0, 0);
        __builtin_prefetch((const void*)(ap1 + 256), 0, 0);

        f4 f0 = __builtin_nontemporal_load((const f4*)(ap0));
        f4 f1 = __builtin_nontemporal_load((const f4*)(ap0 + 4));
        f4 f2 = __builtin_nontemporal_load((const f4*)(ap0 + 16));
        f4 f3 = __builtin_nontemporal_load((const f4*)(ap0 + 20));
        f4 g0 = __builtin_nontemporal_load((const f4*)(ap1));
        f4 g1 = __builtin_nontemporal_load((const f4*)(ap1 + 4));
        f4 g2 = __builtin_nontemporal_load((const f4*)(ap1 + 16));
        f4 g3 = __builtin_nontemporal_load((const f4*)(ap1 + 20));

        Frag a0, a1;
        a0.u[0] = pk2(f0[0], f0[1]); a0.u[1] = pk2(f0[2], f0[3]);
        a0.u[2] = pk2(f1[0], f1[1]); a0.u[3] = pk2(f1[2], f1[3]);
        a0.u[4] = pk2(f2[0], f2[1]); a0.u[5] = pk2(f2[2], f2[3]);
        a0.u[6] = pk2(f3[0], f3[1]); a0.u[7] = pk2(f3[2], f3[3]);
        a1.u[0] = pk2(g0[0], g0[1]); a1.u[1] = pk2(g0[2], g0[3]);
        a1.u[2] = pk2(g1[0], g1[1]); a1.u[3] = pk2(g1[2], g1[3]);
        a1.u[4] = pk2(g2[0], g2[1]); a1.u[5] = pk2(g2[2], g2[3]);
        a1.u[6] = pk2(g3[0], g3[1]); a1.u[7] = pk2(g3[2], g3[3]);

        const unsigned* bbase =
            Bf + (((size_t)kt * ntilesB + (cb << 2)) << 8) + (lane << 3);
#pragma unroll
        for (int t = 0; t < 4; ++t) {
            Frag b;
            const u4* bp = (const u4*)(bbase + ((size_t)t << 8));
            b.q[0] = bp[0];
            b.q[1] = bp[1];
            acc[t] = __builtin_amdgcn_wmma_f32_16x16x32_bf16(
                false, a0.v, false, b.v, (short)0, acc[t], false, false);
            acc[4 + t] = __builtin_amdgcn_wmma_f32_16x16x32_bf16(
                false, a1.v, false, b.v, (short)0, acc[4 + t], false, false);
        }
    }

#pragma unroll
    for (int s = 0; s < 2; ++s) {
#pragma unroll
        for (int t = 0; t < 4; ++t) {
#pragma unroll
            for (int r = 0; r < 8; ++r) {
                float v = acc[s * 4 + t][r];
                if (act) v = lrelu(v);
                C[(size_t)(m0 + s * 16 + r + (half << 3)) * ldc +
                  (size_t)cb * 64 + t * 16 + lm] = v;
            }
        }
    }
}

// ---------------------------------------------------------------------------
// GEMM with both operands pre-swizzled bf16: C = lrelu(Af @ Bf), N = 64.
// Used for lat = lrelu(H^T @ cur_slice): M = 128 (8 mtiles), K = nodes.
// ---------------------------------------------------------------------------
__global__ __launch_bounds__(256) void gemm_sw(const unsigned* __restrict__ Af,
                                               const unsigned* __restrict__ Bf,
                                               float* __restrict__ C,
                                               int nk, int ntilesB, int Mtiles,
                                               int ldc) {
    int wave = blockIdx.x * (blockDim.x >> 5) + (threadIdx.x >> 5);
    int lane = threadIdx.x & 31;
    if (wave >= Mtiles) return;
    int m0   = wave << 4;
    int half = lane >> 4;
    int lm   = lane & 15;

    v8f acc[4] = {};
    for (int kt = 0; kt < nk; ++kt) {
        Frag a;
        const u4* apq =
            (const u4*)(Af + (((size_t)wave * nk + kt) << 8) + (lane << 3));
        a.q[0] = apq[0];
        a.q[1] = apq[1];
        const unsigned* bbase = Bf + (((size_t)kt * ntilesB) << 8) + (lane << 3);
#pragma unroll
        for (int t = 0; t < 4; ++t) {
            Frag b;
            const u4* bp = (const u4*)(bbase + ((size_t)t << 8));
            b.q[0] = bp[0];
            b.q[1] = bp[1];
            acc[t] = __builtin_amdgcn_wmma_f32_16x16x32_bf16(
                false, a.v, false, b.v, (short)0, acc[t], false, false);
        }
    }
#pragma unroll
    for (int t = 0; t < 4; ++t) {
#pragma unroll
        for (int r = 0; r < 8; ++r) {
            C[(size_t)(m0 + r + (half << 3)) * ldc + t * 16 + lm] =
                lrelu(acc[t][r]);
        }
    }
}

// ---------------------------------------------------------------------------
// Elementwise helpers
// ---------------------------------------------------------------------------
__global__ __launch_bounds__(256) void init_cur_acc(const float* __restrict__ iE,
                                                    const float* __restrict__ dE,
                                                    const float* __restrict__ fE,
                                                    float* __restrict__ cur,
                                                    float* __restrict__ acc) {
    int idx = blockIdx.x * blockDim.x + threadIdx.x;
    const int total = N_TOT * LATDIM;
    if (idx >= total) return;
    float v;
    if (idx < ISSUE_N * LATDIM) v = iE[idx];
    else if (idx < (ISSUE_N + DEV_N) * LATDIM) v = dE[idx - ISSUE_N * LATDIM];
    else v = fE[idx - (ISSUE_N + DEV_N) * LATDIM];
    cur[idx] = v;
    acc[idx] = v;
}

__global__ __launch_bounds__(256) void add_update(const float* __restrict__ tem,
                                                  const float* __restrict__ hyp,
                                                  float* __restrict__ cur,
                                                  float* __restrict__ acc) {
    int idx = blockIdx.x * blockDim.x + threadIdx.x;
    if (idx >= N_TOT * LATDIM) return;
    float c = tem[idx] + hyp[idx];
    cur[idx] = c;
    acc[idx] += c;
}

__global__ __launch_bounds__(256) void copy_acc(const float* __restrict__ src,
                                                float* __restrict__ dst) {
    int idx = blockIdx.x * blockDim.x + threadIdx.x;
    if (idx >= N_TOT * LATDIM) return;
    dst[idx] = src[idx];
}

// ---------------------------------------------------------------------------
// Host orchestration
// ---------------------------------------------------------------------------
extern "C" void kernel_launch(void* const* d_in, const int* in_sizes, int n_in,
                              void* d_out, int out_size, void* d_ws, size_t ws_size,
                              hipStream_t stream) {
    (void)in_sizes; (void)n_in; (void)out_size; (void)ws_size;

    const float* adj = (const float*)d_in[0];
    const float* iE  = (const float*)d_in[1];
    const float* dE  = (const float*)d_in[2];
    const float* fE  = (const float*)d_in[3];
    const float* iH  = (const float*)d_in[4];
    const float* dH  = (const float*)d_in[5];
    const float* fH  = (const float*)d_in[6];
    // d_in[7] = keepRate (unused, == 1)

    float* out = (float*)d_out;
    const int NT = N_TOT * LATDIM;            // 1,048,576
    float* out_acc = out;                     // [N, 64]
    float* out_gnn = out + NT;                // [2, N, 64]
    float* out_hyp = out + NT * (1 + LAYERS); // [2, N, 64]

    // --- workspace layout (floats then uints) ---
    float* ws      = (float*)d_ws;
    float* cur     = ws;                        // N*64
    float* acc     = cur + NT;                  // N*64
    float* Hbuf    = acc + NT;                  // N*128
    float* latbuf  = Hbuf + (size_t)N_TOT * HYPER_D;  // 3*128*64
    unsigned* ubase = (unsigned*)(latbuf + 3 * HYPER_D * LATDIM);
    unsigned* curB = ubase;                                      // (N/32)*4*256
    unsigned* hypB = curB + (size_t)(N_TOT / 32) * 4 * 256;      // 3*16*256
    unsigned* latB = hypB + 3 * 16 * 256;                        // 3*16*256
    unsigned* HTA  = latB + 3 * 16 * 256;                        // 4096*256

    const int   nodes[3]  = {ISSUE_N, DEV_N, FILE_N};
    const int   rowoff[3] = {0, ISSUE_N, ISSUE_N + DEV_N};
    const float* emb[3]   = {iE, dE, fE};
    const float* hyp[3]   = {iH, dH, fH};
    size_t htaOff[3];
    htaOff[0] = 0;
    htaOff[1] = (size_t)8 * (ISSUE_N / 32) * 256;
    htaOff[2] = htaOff[1] + (size_t)8 * (DEV_N / 32) * 256;

    const int EB = 256;                 // elementwise block
    const int egrid = (NT + EB - 1) / EB;

    // 1) cur = acc = concat(iE, dE, fE)
    init_cur_acc<<<egrid, EB, 0, stream>>>(iE, dE, fE, cur, acc);

    // 2) repack hyper matrices [64,128] -> B fragments; 3) H_t = emb_t @ hyper_t
    for (int t = 0; t < 3; ++t) {
        unsigned* hb = hypB + (size_t)t * 16 * 256;
        repack_b<<<2, EB, 0, stream>>>(hyp[t], hb, LATDIM, HYPER_D);
        int mt32 = nodes[t] / 32;
        dim3 g((mt32 + 7) / 8, HYPER_D / 64);
        gemm_a32<<<g, EB, 0, stream>>>(emb[t], hb,
                                       Hbuf + (size_t)rowoff[t] * HYPER_D,
                                       nodes[t], LATDIM, LATDIM, HYPER_D,
                                       HYPER_D / 16, /*act=*/0);
    }

    // 4) H^T -> A-layout fragments (once; H is fixed across layers)
    for (int t = 0; t < 3; ++t) {
        int frags = 8 * (nodes[t] / 32);
        repack_at<<<(frags * 32 + EB - 1) / EB, EB, 0, stream>>>(
            Hbuf + (size_t)rowoff[t] * HYPER_D, HTA + htaOff[t], nodes[t]);
    }

    // 5) layers
    for (int layer = 0; layer < LAYERS; ++layer) {
        float* tem = out_gnn + (size_t)layer * NT;
        float* hyo = out_hyp + (size_t)layer * NT;

        // a) cur -> B fragments
        {
            int frags = (N_TOT / 32) * (LATDIM / 16);
            repack_b<<<(frags * 32 + EB - 1) / EB, EB, 0, stream>>>(
                cur, curB, N_TOT, LATDIM);
        }
        // b) tem = lrelu(adj @ cur)  — the HBM-bound streaming GEMM
        {
            dim3 g((N_TOT / 32) / 8, 1);
            gemm_a32<<<g, EB, 0, stream>>>(adj, curB, tem,
                                           N_TOT, N_TOT, N_TOT, LATDIM,
                                           LATDIM / 16, /*act=*/1);
        }
        // c) lat_t = lrelu(H_t^T @ cur_t)  (M=128, K=nodes, N=64)
        for (int t = 0; t < 3; ++t) {
            const unsigned* bSlice = curB + (size_t)(rowoff[t] / 32) * 4 * 256;
            gemm_sw<<<1, EB, 0, stream>>>(HTA + htaOff[t], bSlice,
                                          latbuf + (size_t)t * HYPER_D * LATDIM,
                                          nodes[t] / 32, LATDIM / 16,
                                          HYPER_D / 16, LATDIM);
        }
        // d) lat_t -> B fragments ; e) hyp_t = lrelu(H_t @ lat_t)
        for (int t = 0; t < 3; ++t) {
            unsigned* lb = latB + (size_t)t * 16 * 256;
            repack_b<<<2, EB, 0, stream>>>(latbuf + (size_t)t * HYPER_D * LATDIM,
                                           lb, HYPER_D, LATDIM);
            int mt32 = nodes[t] / 32;
            dim3 g((mt32 + 7) / 8, 1);
            gemm_a32<<<g, EB, 0, stream>>>(Hbuf + (size_t)rowoff[t] * HYPER_D, lb,
                                           hyo + (size_t)rowoff[t] * LATDIM,
                                           nodes[t], HYPER_D, HYPER_D, LATDIM,
                                           LATDIM / 16, /*act=*/1);
        }
        // f) cur = tem + hyp ; acc += cur
        add_update<<<egrid, EB, 0, stream>>>(tem, hyo, cur, acc);
    }

    // 6) emit acc
    copy_acc<<<egrid, EB, 0, stream>>>(acc, out_acc);
}
